// SelfAttentionHead_69449621176480
// MI455X (gfx1250) — matrix operand order repaired and checked by reading, not compiled, not run on hardware
//
#include <hip/hip_runtime.h>

// ---------------------------------------------------------------------------
// Fused causal self-attention head for MI455X (gfx1250, wave32, WMMA bf16).
// Kernel 0: weight convert  (W fp32 -> Wt bf16, transposed [3*64][1024])
// Kernel 1: QKV projection  (double-buffered TDM weights -> 12 WMMAs/slice)
// Kernel 2: flash attention (double-buffered TDM K/V tiles shared by 4 waves)
// ---------------------------------------------------------------------------

#define B_  8
#define T_  2048
#define DM_ 1024
#define DK_ 64
#define WSTR 40   // 80B LDS row stride (64B data + 16B TDM pad): 16B-aligned, conflict-free
#define KSTR 72   // 144B LDS row stride (128B data + 16B TDM pad) for K tiles

typedef __attribute__((ext_vector_type(16))) __bf16   v16bf;
typedef __attribute__((ext_vector_type(2)))  __bf16   v2bf;
typedef __attribute__((ext_vector_type(8)))  float    v8f;
typedef __attribute__((ext_vector_type(4)))  unsigned v4u;
typedef __attribute__((ext_vector_type(8)))  int      v8i_;
typedef __attribute__((ext_vector_type(4)))  int      v4i_;

union V16BF { v16bf v; unsigned u[8]; uint4 q[2]; };

static __device__ __forceinline__ unsigned short f2bf(float f) {
  return __builtin_bit_cast(unsigned short, (__bf16)f);   // native v_cvt, RNE
}
static __device__ __forceinline__ unsigned pack2(float lo, float hi) {
#if __has_builtin(__builtin_amdgcn_cvt_pk_bf16_f32)
  return __builtin_bit_cast(unsigned, __builtin_amdgcn_cvt_pk_bf16_f32(lo, hi));
#else
  v2bf p; p.x = (__bf16)lo; p.y = (__bf16)hi;
  return __builtin_bit_cast(unsigned, p);
#endif
}

typedef __attribute__((address_space(3))) unsigned short lds_us;
static __device__ __forceinline__ unsigned lds_off(unsigned short* p) {
  return (unsigned)(unsigned long long)(lds_us*)p;        // raw LDS byte offset
}

// ---------------------------------------------------------------------------
// Kernel 0: convert W{q,k,v} fp32 [1024,64] -> Wt bf16 transposed [3*64][1024].
// ---------------------------------------------------------------------------
__global__ __launch_bounds__(256) void wconv_kernel(
    const float* __restrict__ Wq, const float* __restrict__ Wk,
    const float* __restrict__ Wv, unsigned short* __restrict__ Wt)
{
  const int e = blockIdx.x * 256 + threadIdx.x;      // [0, 3*65536)
  const int mat = e >> 16;
  const int idx = e & 0xFFFF;
  const int r = idx >> 6, c = idx & 63;
  const float* W = (mat == 0) ? Wq : (mat == 1) ? Wk : Wv;
  Wt[((size_t)(mat * 64 + c) << 10) + r] = f2bf(W[idx]);
}

// ---------------------------------------------------------------------------
// Generic 2D TDM descriptor (2B elements, LDS row padding via pad fields).
// ---------------------------------------------------------------------------
static __device__ __forceinline__ void tdm_load_2d(
    const unsigned short* gsrc, unsigned lds_byte_addr,
    unsigned td0, unsigned td1, unsigned stride0,
    unsigned tile0, unsigned tile1, unsigned padi, unsigned pada)
{
  const unsigned long long ga = (unsigned long long)(size_t)gsrc;
  v4u g0;
  g0[0] = 1u;                                        // count=1, user mode
  g0[1] = lds_byte_addr;                             // lds_addr
  g0[2] = (unsigned)ga;                              // global_addr[31:0]
  g0[3] = (unsigned)(ga >> 32) | (2u << 30);         // global_addr[56:32], type=2
  v8i_ g1;
  g1[0] = (int)((1u << 16)                           // data_size = 2 bytes
              | (1u << 20)                           // pad_enable
              | (padi << 22) | (pada << 25));        // pad interval / amount
  g1[1] = (int)((td0 & 0xFFFFu) << 16);              // tensor_dim0[15:0]
  g1[2] = (int)((td0 >> 16) | ((td1 & 0xFFFFu) << 16));
  g1[3] = (int)((td1 >> 16) | (tile0 << 16));        // tile_dim0
  g1[4] = (int)tile1;                                // tile_dim1, tile_dim2=0
  g1[5] = (int)stride0;                              // tensor_dim0_stride[31:0]
  g1[6] = 0; g1[7] = 0;
  const v4i_ z4 = {0, 0, 0, 0};
  const v8i_ z8 = {0, 0, 0, 0, 0, 0, 0, 0};
  __builtin_amdgcn_tensor_load_to_lds(g0, g1, z4, z4, z8, 0);
}

// ---------------------------------------------------------------------------
// Kernel 1: QKV projection. 4 waves/block, each wave owns 16 rows of x.
// Double-buffered TDM streams 192x32 weight slices under the WMMAs.
// ---------------------------------------------------------------------------
#define WSLICE (3 * DK_ * WSTR)   // ushorts per weight slice buffer

__global__ __launch_bounds__(128) void qkv_proj_kernel(
    const float* __restrict__ x,
    const unsigned short* __restrict__ WtG,
    unsigned short* __restrict__ Qb,
    unsigned short* __restrict__ Kb,
    unsigned short* __restrict__ Vt)
{
  __shared__ unsigned short wt[2][WSLICE];        // 2 x 15360 B

  const int tid  = threadIdx.x;
  const int lane = tid & 31;
  const int wave = tid >> 5;
  const int row0 = blockIdx.x * 64 + wave * 16;   // global row in [0, B*T)
  const int n    = lane & 15;
  const int hi   = lane >> 4;

  v8f accQ[4], accK[4], accV[4];
#pragma unroll
  for (int t = 0; t < 4; ++t) {
    accQ[t] = v8f{0,0,0,0,0,0,0,0};
    accK[t] = v8f{0,0,0,0,0,0,0,0};
    accV[t] = v8f{0,0,0,0,0,0,0,0};
  }

  const float* xrow = &x[(size_t)(row0 + n) * DM_];

  // prologue: DMA slice 0 into buffer 0
  if (tid < 32)
    tdm_load_2d(WtG, lds_off(&wt[0][0]), DM_, 192, DM_, 32, 192, 3, 3);

  for (int kk = 0; kk < DM_; kk += 32) {
    const int buf = (kk >> 5) & 1;
    // issue next slice into other buffer, then wait for current slice only
    if (tid < 32) {
      if (kk + 32 < DM_) {
        tdm_load_2d(WtG + kk + 32, lds_off(&wt[buf ^ 1][0]),
                    DM_, 192, DM_, 32, 192, 3, 3);
        __builtin_amdgcn_s_wait_tensorcnt(1);   // in-order: current buf ready
      } else {
        __builtin_amdgcn_s_wait_tensorcnt(0);
      }
    }
    __syncthreads();

    // --- A regs: 16x32 slice of x (row n), float4 loads + packed cvt -------
    V16BF a;
    {
      const float4 x0 = *(const float4*)&xrow[kk + hi * 8];
      const float4 x1 = *(const float4*)&xrow[kk + hi * 8 + 4];
      const float4 x2 = *(const float4*)&xrow[kk + 16 + hi * 8];
      const float4 x3 = *(const float4*)&xrow[kk + 16 + hi * 8 + 4];
      a.u[0] = pack2(x0.x, x0.y); a.u[1] = pack2(x0.z, x0.w);
      a.u[2] = pack2(x1.x, x1.y); a.u[3] = pack2(x1.z, x1.w);
      a.u[4] = pack2(x2.x, x2.y); a.u[5] = pack2(x2.z, x2.w);
      a.u[6] = pack2(x3.x, x3.y); a.u[7] = pack2(x3.z, x3.w);
    }

    // --- 12 WMMAs, B regs = two ds_load_b128 each --------------------------
#pragma unroll
    for (int mat = 0; mat < 3; ++mat) {
#pragma unroll
      for (int nt = 0; nt < 4; ++nt) {
        unsigned short* basep =
            &wt[buf][(mat * 64 + nt * 16 + n) * WSTR + hi * 8];
        V16BF bm;
        bm.q[0] = *(const uint4*)basep;
        bm.q[1] = *(const uint4*)(basep + 16);
        v8f& acc = (mat == 0) ? accQ[nt] : (mat == 1) ? accK[nt] : accV[nt];
        acc = __builtin_amdgcn_wmma_f32_16x16x32_bf16(
            false, a.v, false, bm.v, (short)0, acc, false, false);
      }
    }
    __syncthreads();   // protect buf before it is DMA-overwritten next+1
  }

  // --- epilogue: Q,K row-major bf16; V transposed [b][dk][t] ---------------
  const int b  = row0 >> 11;
  const int t0 = row0 & (T_ - 1);
#pragma unroll
  for (int nt = 0; nt < 4; ++nt) {
#pragma unroll
    for (int j = 0; j < 8; ++j) {
      const int m  = j + hi * 8;
      const int dk = nt * 16 + n;
      Qb[(size_t)(row0 + m) * DK_ + dk] = f2bf(accQ[nt][j]);
      Kb[(size_t)(row0 + m) * DK_ + dk] = f2bf(accK[nt][j]);
      Vt[((size_t)b * DK_ + dk) * T_ + (t0 + m)] = f2bf(accV[nt][j]);
    }
  }
}

// ---------------------------------------------------------------------------
// Kernel 2: flash attention. Block = 64 query rows (4 waves x 16). K/V tiles
// for each 32-key step are TDM-streamed once per block (double-buffered) and
// shared by all 4 waves; each wave does 4 QK^T + 4 PV WMMAs per step.
// ---------------------------------------------------------------------------
__global__ __launch_bounds__(128) void attn_kernel(
    const unsigned short* __restrict__ Qb,
    const unsigned short* __restrict__ Kb,
    const unsigned short* __restrict__ Vt,
    float* __restrict__ out)
{
  __shared__ unsigned short kt[2][32 * KSTR];     // K tile: 32 keys x 64 dk
  __shared__ unsigned short vt[2][DK_ * WSTR];    // V tile: 64 dk x 32 keys
  __shared__ unsigned short pl[4][16 * WSTR];     // per-wave P bounce (16x32)

  const int tid  = threadIdx.x;
  const int lane = tid & 31;
  const int wave = tid >> 5;
  const int n    = lane & 15;
  const int hi   = lane >> 4;

  const int blocks_per_b = T_ / 64;
  const int b   = blockIdx.x / blocks_per_b;
  const int bm0 = (blockIdx.x % blocks_per_b) * 64;   // block's first query row
  const int m0  = bm0 + wave * 16;                    // this wave's query tile
  const int nend = bm0 + 64;                          // keys needed by block
  const size_t qrow0 = (size_t)b * T_ + m0;
  unsigned short* myP = &pl[wave][0];

  const unsigned short* KbB = &Kb[(size_t)b * T_ * DK_];
  const unsigned short* VtB = &Vt[(size_t)b * DK_ * T_];

  const float NEG_INF = -__builtin_inff();

  // Q A-regs: two 16x32 chunks (dk 0..63), four b128 loads total
  V16BF aq[2];
  {
    const unsigned short* qp = &Qb[(qrow0 + n) * DK_];
#pragma unroll
    for (int c = 0; c < 2; ++c) {
      aq[c].q[0] = *(const uint4*)&qp[c * 32 + hi * 16];
      aq[c].q[1] = *(const uint4*)&qp[c * 32 + hi * 16 + 8];
    }
  }

  float mrun[8], lrun[8];
  v8f o[4];
#pragma unroll
  for (int j = 0; j < 8; ++j) { mrun[j] = NEG_INF; lrun[j] = 0.0f; }
#pragma unroll
  for (int nt = 0; nt < 4; ++nt) o[nt] = v8f{0,0,0,0,0,0,0,0};

  // prologue: DMA first K/V tiles into buffer 0
  if (tid < 32) {
    tdm_load_2d(KbB, lds_off(&kt[0][0]), DK_, T_, DK_, DK_, 32, 4, 3);
    tdm_load_2d(VtB, lds_off(&vt[0][0]), T_, DK_, T_, 32, DK_, 3, 3);
  }

  for (int n0 = 0; n0 < nend; n0 += 32) {
    const int buf = (n0 >> 5) & 1;
    if (tid < 32) {
      if (n0 + 32 < nend) {
        tdm_load_2d(KbB + (size_t)(n0 + 32) * DK_, lds_off(&kt[buf ^ 1][0]),
                    DK_, T_, DK_, DK_, 32, 4, 3);
        tdm_load_2d(VtB + (n0 + 32), lds_off(&vt[buf ^ 1][0]),
                    T_, DK_, T_, 32, DK_, 3, 3);
        __builtin_amdgcn_s_wait_tensorcnt(2);   // in-order: current pair ready
      } else {
        __builtin_amdgcn_s_wait_tensorcnt(0);
      }
    }
    __syncthreads();

    if (n0 < m0 + 16) {   // wave-uniform causal range guard
      // ---- S = Q @ K^T over 32 keys (two 16x16 tiles, dk split 2x32) ------
      v8f s[2];
#pragma unroll
      for (int h = 0; h < 2; ++h) {
        const int kb = n0 + 16 * h;
        unsigned short* kp = &kt[buf][(16 * h + n) * KSTR + hi * 16];
        V16BF bk0, bk1;
        bk0.q[0] = *(const uint4*)kp;
        bk0.q[1] = *(const uint4*)(kp + 8);
        bk1.q[0] = *(const uint4*)(kp + 32);
        bk1.q[1] = *(const uint4*)(kp + 40);
        v8f z = v8f{0,0,0,0,0,0,0,0};
        v8f t = __builtin_amdgcn_wmma_f32_16x16x32_bf16(
            false, aq[0].v, false, bk0.v, (short)0, z, false, false);
        s[h] = __builtin_amdgcn_wmma_f32_16x16x32_bf16(
            false, aq[1].v, false, bk1.v, (short)0, t, false, false);
#pragma unroll
        for (int j = 0; j < 8; ++j)
          if (kb + n > m0 + j + hi * 8) s[h][j] = NEG_INF;  // causal mask
      }

      // ---- online softmax (width-16 butterflies keep rows in-half-wave) ---
      float mnew[8];
#pragma unroll
      for (int j = 0; j < 8; ++j) {
        float v = fmaxf(s[0][j], s[1][j]);
        v = fmaxf(v, __shfl_xor(v, 1, 16));
        v = fmaxf(v, __shfl_xor(v, 2, 16));
        v = fmaxf(v, __shfl_xor(v, 4, 16));
        v = fmaxf(v, __shfl_xor(v, 8, 16));
        mnew[j] = fmaxf(mrun[j], v);
      }
#pragma unroll
      for (int h = 0; h < 2; ++h)
#pragma unroll
        for (int j = 0; j < 8; ++j) {
          const float p = __expf(s[h][j] - mnew[j]);  // exp(-inf)=0 when masked
          s[h][j] = p;
          myP[(j + hi * 8) * WSTR + h * 16 + n] = f2bf(p);
        }
#pragma unroll
      for (int j = 0; j < 8; ++j) {
        float v = s[0][j] + s[1][j];
        v += __shfl_xor(v, 1, 16);
        v += __shfl_xor(v, 2, 16);
        v += __shfl_xor(v, 4, 16);
        v += __shfl_xor(v, 8, 16);
        const float sc = __expf(mrun[j] - mnew[j]);
        lrun[j] = lrun[j] * sc + v;
        mrun[j] = mnew[j];
#pragma unroll
        for (int nt = 0; nt < 4; ++nt) o[nt][j] *= sc;
      }

      // ---- D-layout -> A-layout bounce (in-order DS within the wave) ------
      __builtin_amdgcn_wave_barrier();
      asm volatile("s_wait_dscnt 0" ::: "memory");
      V16BF pa;
      pa.q[0] = *(const uint4*)&myP[n * WSTR + hi * 16];
      pa.q[1] = *(const uint4*)&myP[n * WSTR + hi * 16 + 8];
      __builtin_amdgcn_wave_barrier();

      // ---- O += P @ V (B regs from the shared transposed V tile) ----------
#pragma unroll
      for (int nt = 0; nt < 4; ++nt) {
        unsigned short* vp = &vt[buf][(nt * 16 + n) * WSTR + hi * 16];
        V16BF bv;
        bv.q[0] = *(const uint4*)vp;
        bv.q[1] = *(const uint4*)(vp + 8);
        o[nt] = __builtin_amdgcn_wmma_f32_16x16x32_bf16(
            false, pa.v, false, bv.v, (short)0, o[nt], false, false);
      }
    }
    __syncthreads();   // protect buf before it is DMA-overwritten next+1
  }

  // ---- epilogue: out = O / l ----------------------------------------------
#pragma unroll
  for (int nt = 0; nt < 4; ++nt)
#pragma unroll
    for (int j = 0; j < 8; ++j)
      out[(qrow0 + j + hi * 8) * DK_ + nt * 16 + n] = o[nt][j] / lrun[j];
}

// ---------------------------------------------------------------------------
extern "C" void kernel_launch(void* const* d_in, const int* in_sizes, int n_in,
                              void* d_out, int out_size, void* d_ws, size_t ws_size,
                              hipStream_t stream) {
  (void)in_sizes; (void)n_in; (void)out_size; (void)ws_size;
  const float* x  = (const float*)d_in[0];
  const float* Wk = (const float*)d_in[1];
  const float* Wq = (const float*)d_in[2];
  const float* Wv = (const float*)d_in[3];

  unsigned short* Qb = (unsigned short*)d_ws;                 // [B*T, 64] bf16
  unsigned short* Kb = Qb + (size_t)B_ * T_ * DK_;            // [B*T, 64] bf16
  unsigned short* Vt = Kb + (size_t)B_ * T_ * DK_;            // [B, 64, T] bf16
  unsigned short* Wt = Vt + (size_t)B_ * T_ * DK_;            // [3*64, 1024] bf16

  wconv_kernel<<<(3 * DM_ * DK_) / 256, 256, 0, stream>>>(Wq, Wk, Wv, Wt);
  qkv_proj_kernel<<<(B_ * T_) / 64, 128, 0, stream>>>(x, Wt, Qb, Kb, Vt);
  attn_kernel<<<(B_ * T_) / 64, 128, 0, stream>>>(Qb, Kb, Vt, (float*)d_out);
}
